// BitLinear15b_90134183674153
// MI455X (gfx1250) — 8-bit matrix operand order repaired and checked by reading, compiled, not run on hardware
//
#include <hip/hip_runtime.h>
#include <hip/hip_bf16.h>
#include <stdint.h>


typedef int v8i __attribute__((ext_vector_type(8)));

#define K_DIM 4096
#define M_DIM 8192
#define N_DIM 16384
#define W_ELEMS ((size_t)N_DIM * K_DIM)
#define X_ELEMS ((size_t)M_DIM * K_DIM)
#define RED_BLOCKS 1024

__global__ __launch_bounds__(256)
void bl_abs_partial(const float* __restrict__ w, float* __restrict__ partials) {
    __shared__ float sm[256];
    const float4* w4 = (const float4*)w;
    size_t idx    = (size_t)blockIdx.x * blockDim.x + threadIdx.x;
    size_t stride = (size_t)gridDim.x * blockDim.x;
    float s = 0.f;
    for (size_t i = idx; i < W_ELEMS / 4; i += stride) {
        float4 v = w4[i];
        s += fabsf(v.x) + fabsf(v.y) + fabsf(v.z) + fabsf(v.w);
    }
    sm[threadIdx.x] = s;
    __syncthreads();
    for (int off = 128; off > 0; off >>= 1) {
        if ((int)threadIdx.x < off) sm[threadIdx.x] += sm[threadIdx.x + off];
        __syncthreads();
    }
    if (threadIdx.x == 0) partials[blockIdx.x] = sm[0];
}

__global__ __launch_bounds__(256)
void bl_abs_final(const float* __restrict__ partials, float* __restrict__ sum_out) {
    __shared__ float sm[256];
    float s = 0.f;
    for (int i = threadIdx.x; i < RED_BLOCKS; i += 256) s += partials[i];
    sm[threadIdx.x] = s;
    __syncthreads();
    for (int off = 128; off > 0; off >>= 1) {
        if ((int)threadIdx.x < off) sm[threadIdx.x] += sm[threadIdx.x + off];
        __syncthreads();
    }
    if (threadIdx.x == 0) sum_out[0] = sm[0];
}

__device__ __forceinline__ int8_t bl_tern(float w, float inv_gamma) {
    float m = fminf(rintf(fabsf(w) * inv_gamma), 1.0f);
    int mi = (int)m;
    return (int8_t)((w > 0.f) ? mi : ((w < 0.f) ? -mi : 0));
}

__global__ __launch_bounds__(256)
void bl_quant_w(const float* __restrict__ w, const float* __restrict__ abs_sum,
                int8_t* __restrict__ wq) {
    const float gamma = abs_sum[0] / (float)W_ELEMS + 1e-6f;
    const float inv   = 1.0f / gamma;
    const float4* w4 = (const float4*)w;
    char4* q4 = (char4*)wq;
    size_t idx    = (size_t)blockIdx.x * blockDim.x + threadIdx.x;
    size_t stride = (size_t)gridDim.x * blockDim.x;
    for (size_t i = idx; i < W_ELEMS / 4; i += stride) {
        float4 v = w4[i];
        char4 q;
        q.x = bl_tern(v.x, inv);
        q.y = bl_tern(v.y, inv);
        q.z = bl_tern(v.z, inv);
        q.w = bl_tern(v.w, inv);
        q4[i] = q;
    }
}

__global__ __launch_bounds__(256)
void bl_sign_x(const float* __restrict__ x, int8_t* __restrict__ xs) {
    const float4* x4 = (const float4*)x;
    char4* q4 = (char4*)xs;
    size_t idx    = (size_t)blockIdx.x * blockDim.x + threadIdx.x;
    size_t stride = (size_t)gridDim.x * blockDim.x;
    for (size_t i = idx; i < X_ELEMS / 4; i += stride) {
        float4 v = x4[i];
        char4 q;
        q.x = (int8_t)((v.x > 0.f) - (v.x < 0.f));
        q.y = (int8_t)((v.y > 0.f) - (v.y < 0.f));
        q.z = (int8_t)((v.z > 0.f) - (v.z < 0.f));
        q.w = (int8_t)((v.w > 0.f) - (v.w < 0.f));
        q4[i] = q;
    }
}

__device__ __forceinline__ v8i bl_load_a(const int8_t* p) {
    int2 a0 = *(const int2*)(p);
    int2 a1 = *(const int2*)(p + 16);
    int2 a2 = *(const int2*)(p + 32);
    int2 a3 = *(const int2*)(p + 48);
    return (v8i){a0.x, a0.y, a1.x, a1.y, a2.x, a2.y, a3.x, a3.y};
}
__device__ __forceinline__ v8i bl_load_b(const int8_t* p) {
    int4 b0 = *(const int4*)(p);
    int4 b1 = *(const int4*)(p + 32);
    return (v8i){b0.x, b0.y, b0.z, b0.w, b1.x, b1.y, b1.z, b1.w};
}

#define FRAG_PITCH (16 * K_DIM)

#define CHUNK_LOADS(dstA, dstB, kOff)                                   \
    _Pragma("unroll") for (int i = 0; i < 4; ++i) {                     \
        dstA[i] = bl_load_b(aP + i * FRAG_PITCH + (kOff));              \
        dstB[i] = bl_load_b(bP + i * FRAG_PITCH + (kOff));              \
    }

#define CHUNK_WMMA(aF, bF)                                              \
    _Pragma("unroll") for (int mi = 0; mi < 4; ++mi)                    \
        _Pragma("unroll") for (int ni = 0; ni < 4; ++ni)                \
            acc[mi][ni] = __builtin_amdgcn_wmma_i32_16x16x64_iu8(       \
                true, aF[mi], true, bF[ni], acc[mi][ni], false, false);

__global__ __launch_bounds__(256)
void bl_gemm_tern(const int8_t* __restrict__ A, const int8_t* __restrict__ B,
                  float* __restrict__ C) {
    const int lane   = threadIdx.x & 31;
    const int wave   = threadIdx.x >> 5;
    const int laneLo = lane & 15;
    const int laneHi = lane >> 4;
    const int waveM  = wave >> 2;
    const int waveN  = wave & 3;

    const int mBase = blockIdx.x * 128 + waveM * 64;
    const int nBase = blockIdx.y * 256 + waveN * 64;

    const int8_t* aP = A + (size_t)(mBase + laneLo) * K_DIM + laneHi * 16;
    const int8_t* bP = B + (size_t)(nBase + laneLo) * K_DIM + laneHi * 16;

    v8i acc[4][4];
#pragma unroll
    for (int mi = 0; mi < 4; ++mi)
#pragma unroll
        for (int ni = 0; ni < 4; ++ni)
            acc[mi][ni] = (v8i){0, 0, 0, 0, 0, 0, 0, 0};

    v8i aCur[4], bCur[4], aNxt[4], bNxt[4];
    CHUNK_LOADS(aCur, bCur, 0);

    for (int k0 = 0; k0 < K_DIM - 256; k0 += 256) {
        CHUNK_LOADS(aNxt, bNxt, 64);
        CHUNK_WMMA(aCur, bCur);
        CHUNK_LOADS(aCur, bCur, 128);
        CHUNK_WMMA(aNxt, bNxt);
        CHUNK_LOADS(aNxt, bNxt, 192);
        CHUNK_WMMA(aCur, bCur);
        CHUNK_LOADS(aCur, bCur, 256);
        CHUNK_WMMA(aNxt, bNxt);
        aP += 256;
        bP += 256;
    }
    CHUNK_LOADS(aNxt, bNxt, 64);
    CHUNK_WMMA(aCur, bCur);
    CHUNK_LOADS(aCur, bCur, 128);
    CHUNK_WMMA(aNxt, bNxt);
    CHUNK_LOADS(aNxt, bNxt, 192);
    CHUNK_WMMA(aCur, bCur);
    CHUNK_WMMA(aNxt, bNxt);

    const int rowOff = laneHi * 8;
#pragma unroll
    for (int mi = 0; mi < 4; ++mi) {
#pragma unroll
        for (int ni = 0; ni < 4; ++ni) {
            float* outp = C + (size_t)(mBase + mi * 16 + rowOff) * N_DIM
                            + nBase + ni * 16 + laneLo;
#pragma unroll
            for (int r = 0; r < 8; ++r) {
                __builtin_nontemporal_store((float)acc[mi][ni][r],
                                            outp + (size_t)r * N_DIM);
            }
        }
    }
}

extern "C" void kernel_launch(void* const* d_in, const int* in_sizes, int n_in,
                              void* d_out, int out_size, void* d_ws, size_t ws_size,
                              hipStream_t stream) {
    const float* x = (const float*)d_in[0];
    const float* w = (const float*)d_in[1];
    float* out = (float*)d_out;

    char* ws = (char*)d_ws;
    float*  abs_sum  = (float*)ws;
    float*  partials = (float*)(ws + 256);
    int8_t* xs       = (int8_t*)(ws + 8192);
    int8_t* wq       = (int8_t*)(ws + 8192 + X_ELEMS);

    bl_abs_partial<<<RED_BLOCKS, 256, 0, stream>>>(w, partials);
    bl_abs_final<<<1, 256, 0, stream>>>(partials, abs_sum);
    bl_quant_w<<<16384, 256, 0, stream>>>(w, abs_sum, wq);
    bl_sign_x<<<8192, 256, 0, stream>>>(x, xs);

    dim3 grid(M_DIM / 128, N_DIM / 256);
    bl_gemm_tern<<<grid, 256, 0, stream>>>(xs, wq, out);
}
